// StateCompressor_64424509440485
// MI455X (gfx1250) — matrix operand order repaired and checked by reading, not compiled, not run on hardware
//
#include <hip/hip_runtime.h>
#include <hip/hip_bf16.h>
#include <math.h>

// ---------------------------------------------------------------------------
// Types
// ---------------------------------------------------------------------------
typedef __bf16 bf16_t;
typedef __attribute__((ext_vector_type(16))) __bf16        bf16x16;
typedef __attribute__((ext_vector_type(8)))  float         f32x8;
typedef __attribute__((ext_vector_type(4)))  float         f32x4;
typedef __attribute__((ext_vector_type(4)))  unsigned int  u32x4;

// GCC-style int4 vector matching the async-to-LDS builtin's parameter type
// ('__attribute__((__vector_size__(4 * sizeof(int)))) int') in AS1/AS3.
typedef int v4i_gcc __attribute__((vector_size(16)));
typedef __attribute__((address_space(1))) v4i_gcc as1_v4i;
typedef __attribute__((address_space(3))) v4i_gcc as3_v4i;

union Frag { bf16x16 v; u32x4 q[2]; };

#define LDS_STRIDE 40   // 32 K-elems + 8 pad (80B rows -> 16B aligned, bank spread)

// Async global->LDS copy path (ASYNCcnt), gated on builtin availability.
#if defined(__gfx1250__) && \
    __has_builtin(__builtin_amdgcn_global_load_async_to_lds_b128) && \
    __has_builtin(__builtin_amdgcn_s_wait_asynccnt)
#define USE_ASYNC_LDS 1
#else
#define USE_ASYNC_LDS 0
#endif

__device__ __forceinline__ float gelu_exact(float x) {
  // approximate=False: 0.5*x*(1+erf(x/sqrt(2)))
  return 0.5f * x * (1.0f + erff(x * 0.70710678118654752440f));
}

// ---------------------------------------------------------------------------
// Tiled bf16 WMMA GEMM:  C[M,N] = act(A[M,K] @ B[N,K]^T + bias)
// A, B row-major bf16 (B is torch-style W[out,in]); f32 accumulate.
// Block tile 128x128, K-tile 32, 8 wave32s in 4(M) x 2(N); wave tile 32x64.
// Double-buffered LDS; staging via GLOBAL_LOAD_ASYNC_TO_LDS_B128 when
// available (4 async b128 per thread per tile -> s_wait_asynccnt 4 steady
// state), else synchronous b128 load + ds_store fallback.
// ---------------------------------------------------------------------------
template<int GELU, int WF32, int WBF16>
__global__ __launch_bounds__(256) void gemm_bf16_wmma(
    const bf16_t* __restrict__ A, const bf16_t* __restrict__ Bw,
    const float* __restrict__ bias,
    float* __restrict__ Cf, bf16_t* __restrict__ Cb,
    int M, int Nn, int K)
{
  __shared__ __align__(16) bf16_t As[2][128 * LDS_STRIDE];
  __shared__ __align__(16) bf16_t Bs[2][128 * LDS_STRIDE];

  const int tid   = threadIdx.x;
  const int lane  = tid & 31;
  const int wid   = tid >> 5;
  const int waveM = wid >> 1;        // 0..3 : 32-row slab
  const int waveN = wid & 1;         // 0..1 : 64-col slab
  const int l15   = lane & 15;
  const int g     = lane >> 4;       // half-wave group (0/1)

  const int n0 = blockIdx.x * 128;
  const int m0 = blockIdx.y * 128;

  f32x8 acc[2][4] = {};

  const int nk = K / 32;

  // Stage one 128x32 A-tile + 128x32 B-tile into LDS buffer `buf`.
  auto stage = [&](int buf, int kt) {
    #pragma unroll
    for (int it = 0; it < 2; ++it) {
      int chunk = tid + it * 256;          // 0..511
      int r = chunk >> 2;                  // 0..127
      int c = (chunk & 3) << 3;            // 0,8,16,24
      const bf16_t* ga = &A [(size_t)(m0 + r) * K + kt + c];
      const bf16_t* gb = &Bw[(size_t)(n0 + r) * K + kt + c];
      bf16_t* la = &As[buf][r * LDS_STRIDE + c];
      bf16_t* lb = &Bs[buf][r * LDS_STRIDE + c];
#if USE_ASYNC_LDS
      __builtin_amdgcn_global_load_async_to_lds_b128(
          (as1_v4i*)ga, (as3_v4i*)la, 0, 0);
      __builtin_amdgcn_global_load_async_to_lds_b128(
          (as1_v4i*)gb, (as3_v4i*)lb, 0, 0);
#else
      *(u32x4*)la = *(const u32x4*)ga;
      *(u32x4*)lb = *(const u32x4*)gb;
#endif
    }
  };

  stage(0, 0);

  for (int ki = 0; ki < nk; ++ki) {
    const int cur = ki & 1;

    // Issue next tile's copy into the other buffer (its readers finished at
    // the end-of-body barrier of iteration ki-1).
    if (ki + 1 < nk) stage(cur ^ 1, (ki + 1) * 32);

#if USE_ASYNC_LDS
    // Each stage = 4 async b128 per thread; waiting <=4 drains the older
    // batch (tile ki) while the newer one stays in flight behind the WMMAs.
    if (ki + 1 < nk) __builtin_amdgcn_s_wait_asynccnt(4);
    else             __builtin_amdgcn_s_wait_asynccnt(0);
#endif
    __syncthreads();

    // L2 prefetch two tiles ahead.
    if (ki + 2 < nk) {
      int r = tid >> 2, c = (tid & 3) << 3;
      __builtin_prefetch(&A [(size_t)(m0 + r) * K + (ki + 2) * 32 + c], 0, 1);
      __builtin_prefetch(&Bw[(size_t)(n0 + r) * K + (ki + 2) * 32 + c], 0, 1);
    }

    // ---- load fragments per ISA 7.12.2 VGPR layouts ----
    Frag af[2], bfrag[4];
    #pragma unroll
    for (int mt = 0; mt < 2; ++mt) {
      int r  = waveM * 32 + mt * 16 + l15;
      int ka = g << 3;                 // lanes 0-15: K 0..7 & 16..23; 16-31: 8..15 & 24..31
      af[mt].q[0] = *(const u32x4*)&As[cur][r * LDS_STRIDE + ka];
      af[mt].q[1] = *(const u32x4*)&As[cur][r * LDS_STRIDE + ka + 16];
    }
    #pragma unroll
    for (int nt = 0; nt < 4; ++nt) {
      int r  = waveN * 64 + nt * 16 + l15;
      int kb = g << 3;                 // lanes 0-15: K 0..7 & 16..23; 16-31: 8..15 & 24..31
      bfrag[nt].q[0] = *(const u32x4*)&Bs[cur][r * LDS_STRIDE + kb];
      bfrag[nt].q[1] = *(const u32x4*)&Bs[cur][r * LDS_STRIDE + kb + 16];
    }

    #pragma unroll
    for (int mt = 0; mt < 2; ++mt)
      #pragma unroll
      for (int nt = 0; nt < 4; ++nt)
        acc[mt][nt] = __builtin_amdgcn_wmma_f32_16x16x32_bf16(
            false, af[mt].v, false, bfrag[nt].v,
            (short)0, acc[mt][nt], false, false);

    __syncthreads();   // readers done before buf[cur] is overwritten at ki+1
  }

  // ---- epilogue: bias + (exact) GELU + f32/bf16 writeback ----
  #pragma unroll
  for (int mt = 0; mt < 2; ++mt) {
    #pragma unroll
    for (int nt = 0; nt < 4; ++nt) {
      int gn = n0 + waveN * 64 + nt * 16 + l15;
      float bv = bias ? bias[gn] : 0.0f;
      #pragma unroll
      for (int v = 0; v < 8; ++v) {
        int gm = m0 + waveM * 32 + mt * 16 + v + (g << 3);
        float val = acc[mt][nt][v] + bv;
        if (GELU)  val = gelu_exact(val);
        if (WF32)  Cf[(size_t)gm * Nn + gn] = val;
        if (WBF16) Cb[(size_t)gm * Nn + gn] = (bf16_t)val;
      }
    }
  }
}

// ---------------------------------------------------------------------------
// f32 -> bf16 conversion (vectorized 4-wide; all sizes are multiples of 4)
// ---------------------------------------------------------------------------
__global__ void cvt_f32_to_bf16(const float* __restrict__ in,
                                bf16_t* __restrict__ out, size_t n) {
  size_t i = ((size_t)blockIdx.x * blockDim.x + threadIdx.x) * 4;
  if (i >= n) return;
  f32x4 f = *(const f32x4*)(in + i);
  out[i + 0] = (bf16_t)f[0];
  out[i + 1] = (bf16_t)f[1];
  out[i + 2] = (bf16_t)f[2];
  out[i + 3] = (bf16_t)f[3];
}

// ---------------------------------------------------------------------------
// ||codebook[row]||^2  (one block per row)
// ---------------------------------------------------------------------------
__global__ __launch_bounds__(256) void codebook_norms(
    const float* __restrict__ cb, float* __restrict__ enorm, int Cdim) {
  __shared__ float red[256];
  int row = blockIdx.x;
  float s = 0.f;
  for (int c = threadIdx.x; c < Cdim; c += 256) {
    float v = cb[(size_t)row * Cdim + c];
    s += v * v;
  }
  red[threadIdx.x] = s; __syncthreads();
  for (int w = 128; w > 0; w >>= 1) {
    if (threadIdx.x < w) red[threadIdx.x] += red[threadIdx.x + w];
    __syncthreads();
  }
  if (threadIdx.x == 0) enorm[row] = red[0];
}

// ---------------------------------------------------------------------------
// Per-token argmin over codes (dist = ||e||^2 - 2 x.e; ||x||^2 invariant),
// gather quantized rows (f32 out + bf16 feed), accumulate commitment sum.
// ---------------------------------------------------------------------------
__global__ __launch_bounds__(256) void vq_argmin_gather(
    const float* __restrict__ dots,      // [M, Kcb]
    const float* __restrict__ enorm,     // [Kcb]
    const float* __restrict__ cb,        // [Kcb, Cdim] f32
    const float* __restrict__ comp_f32,  // [M, Cdim]
    float* __restrict__ quant_f32,       // [M, Cdim]
    bf16_t* __restrict__ quant_bf16,     // [M, Cdim]
    int* __restrict__ idx_out,           // [M]
    float* __restrict__ commit_acc,
    int Kcb, int Cdim)
{
  __shared__ float sval[256];
  __shared__ int   sidx[256];
  const int m = blockIdx.x;
  const int t = threadIdx.x;

  float best = 3.402823466e+38f; int bi = Kcb - 1;
  for (int j = t; j < Kcb; j += 256) {        // ascending j -> first-min kept
    float d = enorm[j] - 2.0f * dots[(size_t)m * Kcb + j];
    if (d < best) { best = d; bi = j; }
  }
  sval[t] = best; sidx[t] = bi; __syncthreads();
  for (int w = 128; w > 0; w >>= 1) {
    if (t < w) {
      if (sval[t + w] < sval[t] ||
          (sval[t + w] == sval[t] && sidx[t + w] < sidx[t])) {
        sval[t] = sval[t + w]; sidx[t] = sidx[t + w];
      }
    }
    __syncthreads();
  }
  const int idx = sidx[0];
  __syncthreads();

  float lsum = 0.f;
  for (int c = t; c < Cdim; c += 256) {
    float q = cb[(size_t)idx * Cdim + c];
    quant_f32 [(size_t)m * Cdim + c] = q;
    quant_bf16[(size_t)m * Cdim + c] = (bf16_t)q;
    float diff = comp_f32[(size_t)m * Cdim + c] - q;
    lsum += diff * diff;
  }
  sval[t] = lsum; __syncthreads();
  for (int w = 128; w > 0; w >>= 1) {
    if (t < w) sval[t] += sval[t + w];
    __syncthreads();
  }
  if (t == 0) {
    idx_out[m] = idx;
    atomicAdd(commit_acc, sval[0]);
  }
}

// ---------------------------------------------------------------------------
// sum((a-b)^2) accumulated into *acc (grid-stride)
// ---------------------------------------------------------------------------
__global__ __launch_bounds__(256) void mse_accum(
    const float* __restrict__ a, const float* __restrict__ b,
    float* __restrict__ acc, size_t n) {
  __shared__ float red[256];
  size_t stride = (size_t)gridDim.x * 256;
  float s = 0.f;
  for (size_t i = (size_t)blockIdx.x * 256 + threadIdx.x; i < n; i += stride) {
    float d = a[i] - b[i];
    s += d * d;
  }
  red[threadIdx.x] = s; __syncthreads();
  for (int w = 128; w > 0; w >>= 1) {
    if (threadIdx.x < w) red[threadIdx.x] += red[threadIdx.x + w];
    __syncthreads();
  }
  if (threadIdx.x == 0) atomicAdd(acc, red[0]);
}

__global__ void zero_scalars(float* a) {
  if (threadIdx.x < 2) a[threadIdx.x] = 0.f;
}

__global__ void finalize_losses(const float* __restrict__ acc,
                                float* __restrict__ out_commit,
                                float* __restrict__ out_recon,
                                float cscale, float rscale) {
  *out_commit = acc[0] * cscale;
  *out_recon  = acc[1] * rscale;
}

// ---------------------------------------------------------------------------
// Host-side orchestration
// ---------------------------------------------------------------------------
extern "C" void kernel_launch(void* const* d_in, const int* in_sizes, int n_in,
                              void* d_out, int out_size, void* d_ws, size_t ws_size,
                              hipStream_t stream) {
  (void)in_sizes; (void)n_in; (void)out_size; (void)ws_size;

  const int B = 8, N = 2048, D = 4096, C = 512, Kcb = 1024;
  const int H = D / 2;                 // 2048
  const int M = B * N;                 // 16384 tokens

  const float* states   = (const float*)d_in[0];
  const float* We1      = (const float*)d_in[1];
  const float* be1      = (const float*)d_in[2];
  const float* We2      = (const float*)d_in[3];
  const float* be2      = (const float*)d_in[4];
  const float* Wd1      = (const float*)d_in[5];
  const float* bd1      = (const float*)d_in[6];
  const float* Wd2      = (const float*)d_in[7];
  const float* bd2      = (const float*)d_in[8];
  const float* codebook = (const float*)d_in[9];

  // ---- output layout (tuple, flattened in return order) ----
  float* out_recon = (float*)d_out;                                   // M*D
  float* out_comp  = out_recon + (size_t)M * D;                       // M*C
  float* out_quant = out_comp  + (size_t)M * C;                       // M*C
  int*   out_idx   = (int*)(out_quant + (size_t)M * C);               // M (int32)
  float* out_closs = (float*)(out_idx + M);                           // 1
  float* out_rloss = out_closs + 1;                                   // 1

  // ---- workspace carve (256B-aligned) ----
  char* wsb = (char*)d_ws;
  size_t off = 0;
  auto carve = [&](size_t bytes) -> void* {
    void* p = wsb + off;
    off = (off + bytes + 255) & ~(size_t)255;
    return p;
  };
  float*  ws_acc      = (float*)carve(2 * sizeof(float));     // [commit, recon]
  bf16_t* states_bf   = (bf16_t*)carve((size_t)M * D * 2);
  bf16_t* We1_bf      = (bf16_t*)carve((size_t)H * D * 2);
  bf16_t* We2_bf      = (bf16_t*)carve((size_t)C * H * 2);
  bf16_t* Wd1_bf      = (bf16_t*)carve((size_t)H * C * 2);
  bf16_t* Wd2_bf      = (bf16_t*)carve((size_t)D * H * 2);
  bf16_t* cb_bf       = (bf16_t*)carve((size_t)Kcb * C * 2);
  bf16_t* h_bf        = (bf16_t*)carve((size_t)M * H * 2);
  bf16_t* comp_bf     = (bf16_t*)carve((size_t)M * C * 2);
  bf16_t* quant_bf    = (bf16_t*)carve((size_t)M * C * 2);
  bf16_t* h2_bf       = (bf16_t*)carve((size_t)M * H * 2);
  float*  dots_f32    = (float*)carve((size_t)M * Kcb * 4);
  float*  enorm_f32   = (float*)carve((size_t)Kcb * 4);

  zero_scalars<<<1, 32, 0, stream>>>(ws_acc);

  // ---- f32 -> bf16 staging ----
  auto cvt = [&](const float* src, bf16_t* dst, size_t n) {
    size_t blocks = (n / 4 + 255) / 256;
    cvt_f32_to_bf16<<<dim3((unsigned)blocks), 256, 0, stream>>>(src, dst, n);
  };
  cvt(states,   states_bf, (size_t)M * D);
  cvt(We1,      We1_bf,    (size_t)H * D);
  cvt(We2,      We2_bf,    (size_t)C * H);
  cvt(Wd1,      Wd1_bf,    (size_t)H * C);
  cvt(Wd2,      Wd2_bf,    (size_t)D * H);
  cvt(codebook, cb_bf,     (size_t)Kcb * C);

  // ---- encoder: h = gelu(states @ We1^T + be1)   [M,H] ----
  gemm_bf16_wmma<1, 0, 1><<<dim3(H / 128, M / 128), 256, 0, stream>>>(
      states_bf, We1_bf, be1, nullptr, h_bf, M, H, D);

  // ---- compressed = h @ We2^T + be2              [M,C] (f32 out + bf16 feed) ----
  gemm_bf16_wmma<0, 1, 1><<<dim3(C / 128, M / 128), 256, 0, stream>>>(
      h_bf, We2_bf, be2, out_comp, comp_bf, M, C, H);

  // ---- VQ: dots = compressed @ codebook^T        [M,Kcb] ----
  codebook_norms<<<Kcb, 256, 0, stream>>>(codebook, enorm_f32, C);
  gemm_bf16_wmma<0, 1, 0><<<dim3(Kcb / 128, M / 128), 256, 0, stream>>>(
      comp_bf, cb_bf, nullptr, dots_f32, nullptr, M, Kcb, C);
  vq_argmin_gather<<<M, 256, 0, stream>>>(
      dots_f32, enorm_f32, codebook, out_comp,
      out_quant, quant_bf, out_idx, ws_acc /*commit at [0]*/, Kcb, C);

  // ---- decoder: h2 = gelu(quantized @ Wd1^T + bd1)  [M,H] ----
  gemm_bf16_wmma<1, 0, 1><<<dim3(H / 128, M / 128), 256, 0, stream>>>(
      quant_bf, Wd1_bf, bd1, nullptr, h2_bf, M, H, C);

  // ---- reconstructed = h2 @ Wd2^T + bd2          [M,D] ----
  gemm_bf16_wmma<0, 1, 0><<<dim3(D / 128, M / 128), 256, 0, stream>>>(
      h2_bf, Wd2_bf, bd2, out_recon, nullptr, M, D, H);

  // ---- reconstruction loss ----
  mse_accum<<<4096, 256, 0, stream>>>(out_recon, states, ws_acc + 1,
                                      (size_t)M * D);

  // ---- finalize loss scalars ----
  finalize_losses<<<1, 1, 0, stream>>>(
      ws_acc, out_closs, out_rloss,
      0.25f / ((float)M * (float)C),
      1.0f / ((float)M * (float)D));
}
